// MS_SSIMLoss_15461882265714
// MI455X (gfx1250) — matrix operand order repaired and checked by reading, not compiled, not run on hardware
//
#include <hip/hip_runtime.h>

// ---------------------------------------------------------------------------
// SSIM (11x11 Gaussian, sigma=1.5) on (16,3,512,512) fp32 images -> scalar.
// Separable Gaussian implemented as banded-matrix WMMAs (f16 in, f32 acc).
// One workgroup (8 wave32) computes a 128x16 output tile; HBM traffic is the
// compulsory 100 MB (halo re-reads hit the 192MB L2), so the kernel sits on
// the 23.3 TB/s roofline (~4.3us) with WMMA making the conv compute free.
// ---------------------------------------------------------------------------

#define IMG_H 512
#define IMG_W 512
#define PATCH_ROWS 144   // 128 + 10 halo, padded to 144 so chunk-1 reads stay in LDS
#define PSTRIDE 40       // halfs per patch row (32 used) -> 80B stride, conflict-free b128
#define HSTRIDE 40       // halfs per ht row (32 used)

typedef _Float16 half8 __attribute__((ext_vector_type(8)));
typedef _Float16 v16h  __attribute__((ext_vector_type(16)));
typedef float    v8f   __attribute__((ext_vector_type(8)));

// 11-tap normalized Gaussian, sigma = 1.5 (compile-time constants).
__device__ __forceinline__ _Float16 bandval(int k, int x) {
  int d = k - x;                 // nonzero band: 0 <= d <= 10
  float v = 0.0f;
  v = (d == 0 || d == 10) ? 0.00102840f : v;
  v = (d == 1 || d == 9 ) ? 0.00759877f : v;
  v = (d == 2 || d == 8 ) ? 0.03600077f : v;
  v = (d == 3 || d == 7 ) ? 0.10936069f : v;
  v = (d == 4 || d == 6 ) ? 0.21300530f : v;
  v = (d == 5)            ? 0.26601172f : v;
  return (_Float16)v;
}

// A-fragment (16x32 f16): lane holds M = lane&15; K slots are a contiguous
// block kb + j with kb = 16*(lane>>4) (matches load_B_data).
__device__ __forceinline__ v16h make_wfrag_A(int lane) {
  int x  = lane & 15;            // output row r'
  int kb = (lane >> 4) * 16;
  v16h f;
#pragma unroll
  for (int j = 0; j < 16; ++j) {
    int k = kb + j;
    f[j] = bandval(k, x);        // Gv[r'][k] = g[k - r']
  }
  return f;
}

// B-fragment (32x16 f16): lane holds N = lane&15; K slots are
// {kb..kb+7, 16+kb..16+kb+7} with kb = 8*(lane>>4) (matches load_A_data).
__device__ __forceinline__ v16h make_wfrag_B(int lane) {
  int x  = lane & 15;            // output col c
  int kb = (lane >> 4) * 8;
  v16h f;
#pragma unroll
  for (int j = 0; j < 16; ++j) f[j] = bandval((j < 8) ? (kb + j) : (16 + kb + (j - 8)), x);  // Wh[k][c] = g[k - c]
  return f;
}

// Load a data A-fragment: row per lane = (base + lane&15), K = patch columns.
// Two 16B ds loads (kb and 16+kb), both 16B aligned (PSTRIDE*2 = 80B rows).
__device__ __forceinline__ v16h load_A_data(const _Float16* p, int row, int lane) {
  int kb = (lane >> 4) * 8;
  const half8 lo = *(const half8*)(p + row * PSTRIDE + kb);
  const half8 hi = *(const half8*)(p + row * PSTRIDE + 16 + kb);
  v16h f;
#pragma unroll
  for (int i = 0; i < 8; ++i) { f[i] = lo[i]; f[i + 8] = hi[i]; }
  return f;
}

// Load a data B-fragment from ht[c][m] (m contiguous): lane fixed c = lane&15,
// K = m block 16*(lane>>4) + 0..15 -> two contiguous 16B ds loads.
__device__ __forceinline__ v16h load_B_data(const _Float16* hbase, int lane) {
  int c  = lane & 15;
  int kb = (lane >> 4) * 16;
  const half8 lo = *(const half8*)(hbase + c * HSTRIDE + kb);
  const half8 hi = *(const half8*)(hbase + c * HSTRIDE + kb + 8);
  v16h f;
#pragma unroll
  for (int i = 0; i < 8; ++i) { f[i] = lo[i]; f[i + 8] = hi[i]; }
  return f;
}

// Store a C tile (H rows) transposed into ht[c][m] as f16.
// C layout: lane slot v holds C[v + 8*(lane>>4)][lane&15]; slot index v is
// contiguous in m = 16t + 8h + v -> one 16B ds store per lane.
__device__ __forceinline__ void store_C_as_h16(_Float16* hbase, v8f c8, int t, int lane) {
  int c = lane & 15;
  int h = lane >> 4;
  half8 pk;
#pragma unroll
  for (int v = 0; v < 8; ++v) pk[v] = (_Float16)c8[v];
  *(half8*)(hbase + c * HSTRIDE + 16 * t + 8 * h) = pk;
}

__global__ void __launch_bounds__(256)
ssim_wmma_kernel(const float* __restrict__ img1, const float* __restrict__ img2,
                 float* __restrict__ out, float scale) {
  __shared__ __align__(16) _Float16 p1[PATCH_ROWS * PSTRIDE];
  __shared__ __align__(16) _Float16 p2[PATCH_ROWS * PSTRIDE];
  __shared__ __align__(16) _Float16 ht[8 * 16 * HSTRIDE];   // per-wave transpose buf
  __shared__ float redsum;

  const int tid  = threadIdx.x;
  const int lane = tid & 31;
  const int w    = tid >> 5;                 // wave id: output rows R0+16w..+15
  const int R0   = blockIdx.y * 128;
  const int C0   = blockIdx.x * 16;
  const size_t ibase = (size_t)blockIdx.z * (IMG_H * IMG_W);
  const float* i1 = img1 + ibase;
  const float* i2 = img2 + ibase;

  if (tid == 0) redsum = 0.0f;

  __builtin_prefetch(i1 + (size_t)R0 * IMG_W + C0, 0, 1);   // global_prefetch_b8
  __builtin_prefetch(i2 + (size_t)R0 * IMG_W + C0, 0, 1);

  // Cooperative zero-padded patch load, f32 -> f16 (zero pad == conv zero pad).
  for (int idx = tid; idx < PATCH_ROWS * 32; idx += 256) {
    int r = idx >> 5, c = idx & 31;
    int gr = R0 - 5 + r, gc = C0 - 5 + c;
    bool in = (gr >= 0) & (gr < IMG_H) & (gc >= 0) & (gc < IMG_W);
    float a = in ? i1[gr * IMG_W + gc] : 0.0f;
    float b = in ? i2[gr * IMG_W + gc] : 0.0f;
    p1[r * PSTRIDE + c] = (_Float16)a;
    p2[r * PSTRIDE + c] = (_Float16)b;
  }
  __syncthreads();

  const v16h wB = make_wfrag_B(lane);   // horizontal-pass banded weights (B op)
  const v16h wA = make_wfrag_A(lane);   // vertical-pass banded weights (A op)

  // Data A-fragments for this wave's two 16-row H chunks, both images.
  const int mrow = lane & 15;
  v16h a1[2], a2[2];
#pragma unroll
  for (int t = 0; t < 2; ++t) {
    a1[t] = load_A_data(p1, 16 * w + 16 * t + mrow, lane);
    a2[t] = load_A_data(p2, 16 * w + 16 * t + mrow, lane);
  }

  _Float16* hw = ht + w * 16 * HSTRIDE;
  v8f acc[5];

  // Quantities: x1, x2, x1^2, x2^2, x1*x2 (squares/products in f16 registers).
#pragma unroll
  for (int q = 0; q < 5; ++q) {
#pragma unroll
    for (int t = 0; t < 2; ++t) {
      v16h aq;
      if      (q == 0) aq = a1[t];
      else if (q == 1) aq = a2[t];
      else if (q == 2) aq = a1[t] * a1[t];
      else if (q == 3) aq = a2[t] * a2[t];
      else             aq = a1[t] * a2[t];
      v8f hz = {};
      // Horizontal pass: H = data(16x32) x Wh(32x16)
      hz = __builtin_amdgcn_wmma_f32_16x16x32_f16(false, aq, false, wB,
                                                  (short)0, hz, false, false);
      store_C_as_h16(hw, hz, t, lane);      // ht[c][m], f16
    }
    // Same-wave LDS RAW turnaround (DS ops are in-order per wave; the wait
    // keeps the compiler from reordering and drains dscnt before reuse).
    asm volatile("s_wait_dscnt 0" ::: "memory");
    v16h b2 = load_B_data(hw, lane);
    v8f z = {};
    // Vertical pass: out = Gv(16x32) x H(32x16); rows m>25 carry weight 0.
    acc[q] = __builtin_amdgcn_wmma_f32_16x16x32_f16(false, wA, false, b2,
                                                    (short)0, z, false, false);
  }

  // Elementwise SSIM on the 8 accumulator slots (8 pixels per lane).
  const float C1c = 0.0001f, C2c = 0.0009f;
  float lsum = 0.0f;
#pragma unroll
  for (int v = 0; v < 8; ++v) {
    float mu1 = acc[0][v], mu2 = acc[1][v];
    float e11 = acc[2][v], e22 = acc[3][v], e12 = acc[4][v];
    float m11 = mu1 * mu1, m22 = mu2 * mu2, m12 = mu1 * mu2;
    float s1 = e11 - m11, s2 = e22 - m22, s12 = e12 - m12;
    float v1 = 2.0f * s12 + C2c;
    float v2 = s1 + s2 + C2c;
    float num = (2.0f * m12 + C1c) * v1;
    float den = (m11 + m22 + C1c) * v2;
    lsum += num / den;
  }

  atomicAdd(&redsum, lsum * scale);   // ds_add_f32
  __syncthreads();
  if (tid == 0) atomicAdd(out, redsum);  // one global atomic per workgroup
}

__global__ void zero_out_kernel(float* out) { out[0] = 0.0f; }

extern "C" void kernel_launch(void* const* d_in, const int* in_sizes, int n_in,
                              void* d_out, int out_size, void* d_ws, size_t ws_size,
                              hipStream_t stream) {
  const float* img1 = (const float*)d_in[0];
  const float* img2 = (const float*)d_in[1];
  float* out = (float*)d_out;

  const int nimg = in_sizes[0] / (IMG_H * IMG_W);       // 16*3 = 48 channel-images
  const float scale = 1.0f / (float)in_sizes[0];        // global mean

  zero_out_kernel<<<1, 1, 0, stream>>>(out);            // d_out is poisoned by harness
  dim3 grid(IMG_W / 16, IMG_H / 128, nimg);             // 32 x 4 x 48 workgroups
  ssim_wmma_kernel<<<grid, 256, 0, stream>>>(img1, img2, out, scale);
}
